// CausalSelfAttention_28467043237962
// MI455X (gfx1250) — matrix operand order repaired and checked by reading, not compiled, not run on hardware
//
#include <hip/hip_runtime.h>
#include <hip/hip_bf16.h>

typedef __bf16 bf16_t;
typedef __attribute__((ext_vector_type(16))) __bf16 v16bf;
typedef __attribute__((ext_vector_type(8)))  __bf16 v8bf;
typedef __attribute__((ext_vector_type(4)))  __bf16 v4bf;
typedef __attribute__((ext_vector_type(8)))  float  v8f;

#define DEV static __device__ __forceinline__

static constexpr int Bc   = 2;
static constexpr int Tc   = 2048;
static constexpr int DIMc = 1024;
static constexpr int Hc   = 16;
static constexpr int HDc  = 64;
static constexpr float SCALEc = 0.12f;
// ln(1024)/15  (linspace(0,1,16) exponent step for rope inv_freq)
static constexpr float ROPE_LN_STEP = 0.46209812037329693f;

DEV v8f wmma_bf16(v16bf a, v16bf b, v8f c) {
    return __builtin_amdgcn_wmma_f32_16x16x32_bf16(false, a, false, b, (short)0, c, false, false);
}

// ---- WMMA fragment loaders (layouts per CDNA5 ISA 7.12.2, wave32) ----
// A (16x32, bf16): lane m = lane&15; lanes<16 hold K {kb..kb+7, kb+16..kb+23},
// lanes>=16 hold K {kb+8..kb+15, kb+24..kb+31}; regs packed 2 halves/VGPR.
DEV v16bf ld_a_bf16(const bf16_t* __restrict__ base, int ld, int lane, int kb) {
    const int m  = lane & 15;
    const int kk = kb + ((lane & 16) ? 8 : 0);
    const bf16_t* p = base + (size_t)m * ld + kk;
    v8bf lo = *(const v8bf*)p;
    v8bf hi = *(const v8bf*)(p + 16);
    v16bf a;
#pragma unroll
    for (int i = 0; i < 8; ++i) { a[i] = lo[i]; a[8 + i] = hi[i]; }
    return a;
}
// B (32x16, bf16): lane n = lane&15; lanes<16 hold K {kb..kb+7, kb+16..kb+23},
// lanes>=16 hold K {kb+8..kb+15, kb+24..kb+31} (same order as A).
DEV v16bf ld_b_bf16(const bf16_t* __restrict__ base, int ld, int lane, int kb) {
    const int n  = lane & 15;
    const int kk = kb + ((lane & 16) ? 8 : 0);
    const bf16_t* p = base + (size_t)n * ld + kk;
    v8bf lo = *(const v8bf*)p;
    v8bf hi = *(const v8bf*)(p + 16);
    v16bf b;
#pragma unroll
    for (int i = 0; i < 8; ++i) { b[i] = lo[i]; b[8 + i] = hi[i]; }
    return b;
}

// ============================================================================
// Kernel 0: fp32 -> bf16 pre-conversion (x and the 4*DIM x DIM weight tensor).
// Converting once removes the per-GEMM-tile v_cvt chains and halves L2 bytes.
// ============================================================================
__global__ void __launch_bounds__(256) cvt_bf16_kernel(const float* __restrict__ in,
                                                       bf16_t* __restrict__ out, int n4) {
    const int i = blockIdx.x * blockDim.x + threadIdx.x;
    if (i >= n4) return;
    const float4 v = ((const float4*)in)[i];
    v4bf o; o[0] = (__bf16)v.x; o[1] = (__bf16)v.y; o[2] = (__bf16)v.z; o[3] = (__bf16)v.w;
    *(v4bf*)(out + (size_t)i * 4) = o;
}

// ============================================================================
// Generic bf16 GEMM:  C[m,n] = sum_k A[m,k] * B[n,k]   (A,B row-major over k,
// ld = DIMc = 1024).  Wave tile 64x64: 4 A-frags x 4 B-frags -> 16 WMMAs per
// 32-wide K step, 128 accumulator VGPRs.  All 16 b128 loads of a K step are
// issued up-front so they clause and overlap the previous step's WMMA tail.
// ============================================================================
__global__ void __launch_bounds__(128) gemm_bf16_kernel(const bf16_t* __restrict__ A,
                                                        const bf16_t* __restrict__ B,
                                                        float* __restrict__ C,
                                                        int nGroups, int ldc) {
    const int lane = threadIdx.x & 31;
    const int wave = threadIdx.x >> 5;
    const int mt = blockIdx.x / nGroups;
    const int cg = blockIdx.x % nGroups;
    const int m0 = mt * 64;
    const int n0 = cg * 256 + wave * 64;
    const bf16_t* Ab = A + (size_t)m0 * DIMc;
    const bf16_t* Bb = B + (size_t)n0 * DIMc;
    v8f acc[4][4] = {};
    for (int k = 0; k < DIMc; k += 32) {
        // unconditional speculative prefetch of the next K step (gfx1250
        // global_prefetch_b8; translation faults / tail overshoot are dropped)
        __builtin_prefetch((const void*)(Ab + ((size_t)(lane & 15)) * DIMc + k + 32), 0, 1);
        __builtin_prefetch((const void*)(Bb + ((size_t)(lane & 15)) * DIMc + k + 32), 0, 1);
        v16bf av[4], bv[4];
#pragma unroll
        for (int mi = 0; mi < 4; ++mi)
            av[mi] = ld_a_bf16(Ab + (size_t)(mi * 16) * DIMc, DIMc, lane, k);
#pragma unroll
        for (int j = 0; j < 4; ++j)
            bv[j] = ld_b_bf16(Bb + (size_t)(j * 16) * DIMc, DIMc, lane, k);
#pragma unroll
        for (int mi = 0; mi < 4; ++mi)
#pragma unroll
            for (int j = 0; j < 4; ++j)
                acc[mi][j] = wmma_bf16(av[mi], bv[j], acc[mi][j]);
    }
    // C layout: lanes<16 -> rows 0..7 col=lane; lanes>=16 -> rows 8..15 col=lane-16
    const int n  = lane & 15;
    const int r0 = (lane & 16) ? 8 : 0;
#pragma unroll
    for (int mi = 0; mi < 4; ++mi)
#pragma unroll
        for (int j = 0; j < 4; ++j)
#pragma unroll
            for (int i = 0; i < 8; ++i)
                C[(size_t)(m0 + mi * 16 + r0 + i) * ldc + (n0 + j * 16 + n)] = acc[mi][j][i];
}

// ============================================================================
// Kernel 2: per-head RMS norm + RoPE (q,k) + lambda blend (v); writes bf16
// q,k as [b*H+h][t][d] and v TRANSPOSED as [b*H+h][d][t].
// ============================================================================
__global__ void __launch_bounds__(256) qkv_post_kernel(const float* __restrict__ qkv,
                                                       const float* __restrict__ ve,
                                                       const float* __restrict__ lambdas,
                                                       bf16_t* __restrict__ qb,
                                                       bf16_t* __restrict__ kb,
                                                       bf16_t* __restrict__ vt) {
    const int idx = blockIdx.x * blockDim.x + threadIdx.x;  // (b*T+t)*H + h
    if (idx >= Bc * Tc * Hc) return;
    const int h  = idx & (Hc - 1);
    const int bt = idx / Hc;
    const int t  = bt & (Tc - 1);
    const int b  = bt / Tc;
    const float* qr = qkv + (size_t)bt * 3072 + h * HDc;
    const float* kr = qr + DIMc;
    const float* vr = qr + 2 * DIMc;
    float sq = 0.f, sk = 0.f, sv = 0.f;
#pragma unroll 8
    for (int d = 0; d < HDc; ++d) {
        float a = qr[d], bb = kr[d], c = vr[d];
        sq += a * a; sk += bb * bb; sv += c * c;
    }
    const float rq = rsqrtf(sq * (1.0f / HDc) + 1e-6f);
    const float rk = rsqrtf(sk * (1.0f / HDc) + 1e-6f);
    const float rv = rsqrtf(sv * (1.0f / HDc) + 1e-6f);
    const size_t rowo = ((size_t)(b * Hc + h) * Tc + t) * HDc;
#pragma unroll 4
    for (int j = 0; j < 32; ++j) {
        float inv = (j < 16) ? __expf(-ROPE_LN_STEP * (float)j) : 0.0f;
        float s, c;
        __sincosf((float)t * inv, &s, &c);
        float q1 = qr[j] * rq, q2 = qr[j + 32] * rq;
        float k1 = kr[j] * rk, k2 = kr[j + 32] * rk;
        qb[rowo + j]      = (bf16_t)( q1 * c + q2 * s);
        qb[rowo + 32 + j] = (bf16_t)(-q1 * s + q2 * c);
        kb[rowo + j]      = (bf16_t)( k1 * c + k2 * s);
        kb[rowo + 32 + j] = (bf16_t)(-k1 * s + k2 * c);
    }
    const float l0 = lambdas[0], l1 = lambdas[1];
    const float* vep = ve + (size_t)bt * DIMc + h * HDc;
    const size_t vbase = ((size_t)(b * Hc + h) * HDc) * Tc + t;
#pragma unroll 8
    for (int d = 0; d < HDc; ++d)
        vt[vbase + (size_t)d * Tc] = (bf16_t)(vr[d] * rv * l0 + l1 * vep[d]);
}

// ============================================================================
// Kernel 3: flash attention, one wave per (b,h,16-query tile).
// Computes S^T = K·Q^T so the two 16-key f32 C tiles map bit-exactly onto the
// bf16 A-fragment layout of P (16 queries x 32 keys) -- no cross-lane shuffle
// of P needed.  Online softmax over 32-key steps; O += P·V via V^T loads.
// ============================================================================
__global__ void __launch_bounds__(32) attn_kernel(const bf16_t* __restrict__ qb,
                                                  const bf16_t* __restrict__ kb,
                                                  const bf16_t* __restrict__ vt,
                                                  bf16_t* __restrict__ yb) {
    const int lane = threadIdx.x & 31;
    const int qt = blockIdx.x & 127;      // T/16 = 128 query tiles
    const int bh = blockIdx.x >> 7;
    const int b = bh >> 4, h = bh & 15;
    const int qbase = qt * 16;
    const bf16_t* Qp = qb + ((size_t)bh * Tc + qbase) * HDc;
    const bf16_t* Kp = kb + (size_t)bh * Tc * HDc;
    const bf16_t* Vt = vt + (size_t)bh * HDc * Tc;   // [d][t]
    // Q as B-operand of S^T = K·Q^T (n = query, K = d), loaded once.
    const v16bf q0 = ld_b_bf16(Qp, HDc, lane, 0);
    const v16bf q1 = ld_b_bf16(Qp, HDc, lane, 32);
    v8f o[4] = {};
    float mrow = -3.0e38f, lrow = 0.f;
    const int q_abs = qbase + (lane & 15);
    const int r0 = (lane & 16) ? 8 : 0;    // key sub-offset AND O-row offset

    for (int s0 = 0; s0 < qbase + 16; s0 += 32) {
        // speculative prefetch of the next 32-key K tile and V^T columns;
        // overlaps the softmax VALU tail of this iteration.
        __builtin_prefetch((const void*)(Kp + (size_t)(s0 + 32 + lane) * HDc), 0, 1);
        __builtin_prefetch((const void*)(Vt + (size_t)(lane * 2) * Tc + s0 + 32), 0, 1);
        v8f c0 = {}, c1 = {};
        {
            v16bf ka = ld_a_bf16(Kp + (size_t)s0 * HDc, HDc, lane, 0);
            c0 = wmma_bf16(ka, q0, c0);
            ka = ld_a_bf16(Kp + (size_t)s0 * HDc, HDc, lane, 32);
            c0 = wmma_bf16(ka, q1, c0);
            ka = ld_a_bf16(Kp + (size_t)(s0 + 16) * HDc, HDc, lane, 0);
            c1 = wmma_bf16(ka, q0, c1);
            ka = ld_a_bf16(Kp + (size_t)(s0 + 16) * HDc, HDc, lane, 32);
            c1 = wmma_bf16(ka, q1, c1);
        }
        // scale + causal mask + tile max (keys split: this lane holds r0..r0+7 / +16)
        float sv0[8], sv1[8];
        float mnew = mrow;
#pragma unroll
        for (int i = 0; i < 8; ++i) {
            int k0 = s0 + r0 + i;
            float f0 = c0[i] * SCALEc + ((k0 > q_abs) ? -1.0e9f : 0.0f);
            sv0[i] = f0; mnew = fmaxf(mnew, f0);
            int k1 = s0 + 16 + r0 + i;
            float f1 = c1[i] * SCALEc + ((k1 > q_abs) ? -1.0e9f : 0.0f);
            sv1[i] = f1; mnew = fmaxf(mnew, f1);
        }
        mnew = fmaxf(mnew, __shfl_xor(mnew, 16, 32)); // other key-half, same query
        const float alpha = __expf(mrow - mnew);
        mrow = mnew;
        // P in A-fragment layout directly from the two C accumulators
        v16bf pa; float ls = 0.f;
#pragma unroll
        for (int i = 0; i < 8; ++i) {
            float p0 = __expf(sv0[i] - mnew);
            float p1 = __expf(sv1[i] - mnew);
            pa[i] = (bf16_t)p0; pa[8 + i] = (bf16_t)p1;
            ls += p0 + p1;
        }
        ls += __shfl_xor(ls, 16, 32);
        lrow = lrow * alpha + ls;
        // rescale O: row r0+i needs alpha of query r0+i (lives in lane (r0+i))
        float ar[8];
#pragma unroll
        for (int i = 0; i < 8; ++i) ar[i] = __shfl(alpha, r0 + i, 32);
#pragma unroll
        for (int j = 0; j < 4; ++j)
#pragma unroll
            for (int i = 0; i < 8; ++i) o[j][i] *= ar[i];
        // O += P·V ; B-operand: lane n = d, K = keys, contiguous in V^T rows
#pragma unroll
        for (int j = 0; j < 4; ++j) {
            const int d = j * 16 + (lane & 15);
            const bf16_t* vp = Vt + (size_t)d * Tc + s0 + ((lane & 16) ? 8 : 0);
            v8bf vlo = *(const v8bf*)vp;
            v8bf vhi = *(const v8bf*)(vp + 16);
            v16bf vb;
#pragma unroll
            for (int i = 0; i < 8; ++i) { vb[i] = vlo[i]; vb[8 + i] = vhi[i]; }
            o[j] = wmma_bf16(pa, vb, o[j]);
        }
    }
    // normalize rows and store y[b, t, h*64 + d] as bf16 for the out-projection
    const float inv = 1.0f / lrow;
    float ri[8];
#pragma unroll
    for (int i = 0; i < 8; ++i) ri[i] = __shfl(inv, r0 + i, 32);
    const int n = lane & 15;
#pragma unroll
    for (int j = 0; j < 4; ++j)
#pragma unroll
        for (int i = 0; i < 8; ++i) {
            int t = qbase + r0 + i;
            yb[(size_t)(b * Tc + t) * DIMc + h * HDc + j * 16 + n] =
                (bf16_t)(o[j][i] * ri[i]);
        }
}

extern "C" void kernel_launch(void* const* d_in, const int* in_sizes, int n_in,
                              void* d_out, int out_size, void* d_ws, size_t ws_size,
                              hipStream_t stream) {
    const float* x       = (const float*)d_in[0];
    const float* ve      = (const float*)d_in[1];
    const float* lambdas = (const float*)d_in[2];
    // d_in[3] = attn_mask: pure causal -1e9 mask, applied analytically in-kernel
    const float* qkvo    = (const float*)d_in[4];
    float* out = (float*)d_out;

    char* ws = (char*)d_ws;
    const size_t QKV_BYTES = (size_t)Bc * Tc * 3 * DIMc * sizeof(float);   // 50331648
    const size_t HT_BYTES  = (size_t)Bc * Hc * Tc * HDc * sizeof(bf16_t);  // 8388608
    const size_t XB_BYTES  = (size_t)Bc * Tc * DIMc * sizeof(bf16_t);      // 8388608
    float*  qkv = (float*)ws;
    bf16_t* qb  = (bf16_t*)(ws + QKV_BYTES);
    bf16_t* kb  = (bf16_t*)(ws + QKV_BYTES + 1 * HT_BYTES);
    bf16_t* vt  = (bf16_t*)(ws + QKV_BYTES + 2 * HT_BYTES);
    bf16_t* yb  = (bf16_t*)(ws + QKV_BYTES + 3 * HT_BYTES);
    bf16_t* xb  = (bf16_t*)(ws + QKV_BYTES + 4 * HT_BYTES);
    bf16_t* wb  = (bf16_t*)(ws + QKV_BYTES + 4 * HT_BYTES + XB_BYTES);
    (void)ws_size; (void)in_sizes; (void)n_in; (void)out_size;

    // 0) one-time fp32 -> bf16 packing of activations and all 4 weight matrices
    {
        const int nx4 = (Bc * Tc * DIMc) / 4;          // 1,048,576
        const int nw4 = (4 * DIMc * DIMc) / 4;         // 1,048,576
        cvt_bf16_kernel<<<(nx4 + 255) / 256, 256, 0, stream>>>(x, xb, nx4);
        cvt_bf16_kernel<<<(nw4 + 255) / 256, 256, 0, stream>>>(qkvo, wb, nw4);
    }
    // 1) QKV projection: M=4096, N=3072  (64x256 per block, 64x64 per wave)
    gemm_bf16_kernel<<<(Bc * Tc / 64) * (3 * DIMc / 256), 128, 0, stream>>>(
        xb, wb, qkv, 3 * DIMc / 256, 3 * DIMc);
    // 2) norm + rope + blend
    qkv_post_kernel<<<(Bc * Tc * Hc) / 256, 256, 0, stream>>>(qkv, ve, lambdas, qb, kb, vt);
    // 3) flash attention
    attn_kernel<<<Bc * Hc * (Tc / 16), 32, 0, stream>>>(qb, kb, vt, yb);
    // 4) output projection: M=4096, N=1024
    gemm_bf16_kernel<<<(Bc * Tc / 64) * (DIMc / 256), 128, 0, stream>>>(
        yb, wb + (size_t)3 * DIMc * DIMc, out, DIMc / 256, DIMc);
}
